// SOCPooling_11690900979872
// MI455X (gfx1250) — matrix operand order repaired and checked by reading, not compiled, not run on hardware
//
#include <hip/hip_runtime.h>

#define SPD 24
#define WPAD 32            // W planes padded to 32 rows (24..31 zero)
#define TDIM 1024
#define DDIM 768
#define BATCH 128
#define NPART 8            // T-partitions per batch in the main kernel (grid.x)
#define OUT_TRI (SPD*(SPD+1)/2)   // 300
#define ACC_STRIDE (OUT_TRI + SPD) // 300 (upper-tri G partial) + 24 (col sums)

typedef __attribute__((ext_vector_type(16))) __bf16 v16bf;
typedef __attribute__((ext_vector_type(8)))  float  v8f;
typedef __attribute__((ext_vector_type(4)))  float  f32x4;
typedef __attribute__((ext_vector_type(4)))  unsigned u32x4;

union BF16Frag {
  v16bf v;
  unsigned u[8];
  u32x4 q[2];
};

// map linear upper-triangle index (row-major triu) -> (i,j)
__device__ __forceinline__ void triu_ij(int t, int& i, int& j) {
  int ii = 0, rem = t;
  while (rem >= SPD - ii) { rem -= SPD - ii; ++ii; }
  i = ii; j = ii + rem;
}

// ---------------------------------------------------------------------------
// Kernel 1: convert W [24,768] f32 into packed bf16 hi / lo residual planes,
// padded to 32 rows (rows 24..31 = 0 so B-fragment loads are unconditional).
// Layout: row-major, 2 bf16 per uint -> [32][384] uints per plane.
// ---------------------------------------------------------------------------
__global__ void soc_prep_w(const float* __restrict__ W,
                           unsigned* __restrict__ Whi,
                           unsigned* __restrict__ Wlo) {
  int t = blockIdx.x * 256 + threadIdx.x;
  if (t < WPAD * (DDIM / 2)) {
    unsigned uh = 0u, ul = 0u;
    if (t < SPD * (DDIM / 2)) {
      union { __bf16 b[2]; unsigned u; } ph, pl;
#pragma unroll
      for (int j = 0; j < 2; ++j) {
        float f = W[2 * t + j];
        __bf16 h = (__bf16)f;            // RNE
        ph.b[j] = h;
        pl.b[j] = (__bf16)(f - (float)h);
      }
      uh = ph.u; ul = pl.u;
    }
    Whi[t] = uh;
    Wlo[t] = ul;
  }
}

// ---------------------------------------------------------------------------
// Kernel 2: p = x W^T via split-bf16 WMMA (3 terms ~ f32 accuracy), then
// per-block partial upper-tri G = p^T p and s = sum_t p into workspace
// (deterministic, no atomics). grid = (NPART, BATCH), block = 256 (8 waves).
// Each wave: one 16-row T tile x 24 cols, K swept in steps of 32.
// ---------------------------------------------------------------------------
__global__ __launch_bounds__(256) void soc_proj_cov(
    const float* __restrict__ x,
    const unsigned* __restrict__ Whi,
    const unsigned* __restrict__ Wlo,
    float* __restrict__ acc) {
  __shared__ float ptile[8][16][25];   // [wave][row][col(24, padded)]

  const int tid  = threadIdx.x;
  const int lane = tid & 31;
  const int wv   = tid >> 5;
  const int b    = blockIdx.y;
  const int t0   = blockIdx.x * 128 + wv * 16;
  const int m    = lane & 15;          // A: row of tile this lane holds
  const int hi   = lane >> 4;          // lane half selects K sub-chunks
  const int nA   = lane & 15;          // B/D: column this lane holds

  const float* xrow = x + ((size_t)b * TDIM + t0 + m) * DDIM;
  const unsigned* wh0 = Whi + nA * (DDIM / 2);
  const unsigned* wl0 = Wlo + nA * (DDIM / 2);
  const unsigned* wh1 = Whi + (16 + nA) * (DDIM / 2);   // rows 24..31 are zero
  const unsigned* wl1 = Wlo + (16 + nA) * (DDIM / 2);

  v8f c0, c1;
#pragma unroll
  for (int i = 0; i < 8; ++i) { c0[i] = 0.f; c1[i] = 0.f; }

  for (int k0 = 0; k0 < DDIM; k0 += 32) {
    // A fragment: lane<16 holds K k0..k0+15, lane>=16 holds K k0+16..k0+31
    const int kb = k0 + (hi << 4);
    f32x4 rv[4];
    rv[0] = __builtin_nontemporal_load((const f32x4*)(xrow + kb));
    rv[1] = __builtin_nontemporal_load((const f32x4*)(xrow + kb + 4));
    rv[2] = __builtin_nontemporal_load((const f32x4*)(xrow + kb + 8));
    rv[3] = __builtin_nontemporal_load((const f32x4*)(xrow + kb + 12));
    BF16Frag ah, al;
#pragma unroll
    for (int j = 0; j < 16; ++j) {
      float f = rv[j >> 2][j & 3];
      __bf16 h = (__bf16)f;                  // v_cvt_pk_bf16_f32, RNE
      ah.v[j] = h;
      al.v[j] = (__bf16)(f - (float)h);      // v_fma_mix residual
    }
    // B fragments: lane<16 holds K k0..k0+15, lane>=16 holds K k0+16..k0+31
    const int kb2 = (k0 >> 1) + (hi << 3);
    BF16Frag b0h, b0l, b1h, b1l;
    b0h.q[0] = *(const u32x4*)(wh0 + kb2);
    b0h.q[1] = *(const u32x4*)(wh0 + kb2 + 4);
    b0l.q[0] = *(const u32x4*)(wl0 + kb2);
    b0l.q[1] = *(const u32x4*)(wl0 + kb2 + 4);
    b1h.q[0] = *(const u32x4*)(wh1 + kb2);
    b1h.q[1] = *(const u32x4*)(wh1 + kb2 + 4);
    b1l.q[0] = *(const u32x4*)(wl1 + kb2);
    b1l.q[1] = *(const u32x4*)(wl1 + kb2 + 4);
    // p ~= x_hi*W_hi + x_hi*W_lo + x_lo*W_hi  (drop lo*lo; error ~2^-16)
    c0 = __builtin_amdgcn_wmma_f32_16x16x32_bf16(false, ah.v, false, b0h.v, (short)0, c0, false, false);
    c0 = __builtin_amdgcn_wmma_f32_16x16x32_bf16(false, ah.v, false, b0l.v, (short)0, c0, false, false);
    c0 = __builtin_amdgcn_wmma_f32_16x16x32_bf16(false, al.v, false, b0h.v, (short)0, c0, false, false);
    c1 = __builtin_amdgcn_wmma_f32_16x16x32_bf16(false, ah.v, false, b1h.v, (short)0, c1, false, false);
    c1 = __builtin_amdgcn_wmma_f32_16x16x32_bf16(false, ah.v, false, b1l.v, (short)0, c1, false, false);
    c1 = __builtin_amdgcn_wmma_f32_16x16x32_bf16(false, al.v, false, b1h.v, (short)0, c1, false, false);
  }

  // D layout (lane half -> M=v or v+8, lane&15 -> N) into LDS tile
#pragma unroll
  for (int v = 0; v < 8; ++v) {
    int mm = v + (hi << 3);
    ptile[wv][mm][nA] = c0[v];
    if (nA < 8) ptile[wv][mm][16 + nA] = c1[v];
  }
  __syncthreads();

  // Block-wide partial G (upper triangle only) and s over 128 T rows.
  float* accb = acc + ((size_t)b * NPART + blockIdx.x) * ACC_STRIDE;
  for (int e = tid; e < OUT_TRI; e += 256) {
    int k, l;
    triu_ij(e, k, l);
    float sum = 0.f;
    for (int w2 = 0; w2 < 8; ++w2)
#pragma unroll 4
      for (int m2 = 0; m2 < 16; ++m2)
        sum += ptile[w2][m2][k] * ptile[w2][m2][l];
    accb[e] = sum;
  }
  if (tid < SPD) {
    float sum = 0.f;
    for (int w2 = 0; w2 < 8; ++w2)
      for (int m2 = 0; m2 < 16; ++m2)
        sum += ptile[w2][m2][tid];
    accb[OUT_TRI + tid] = sum;
  }
}

// ---------------------------------------------------------------------------
// Kernel 3: per-batch finalize. cov (exactly symmetric by construction) ->
// trace-norm + eps*I -> parallel cyclic Jacobi (12 disjoint rotations/round)
// -> V diag(log lambda) V^T -> upper triangle. grid = 128, block = 288.
// ---------------------------------------------------------------------------
__global__ __launch_bounds__(288) void soc_finalize(const float* __restrict__ acc,
                                                    float* __restrict__ out) {
  __shared__ float A[SPD][SPD + 1];
  __shared__ float V[SPD][SPD + 1];
  __shared__ float sv[SPD];
  __shared__ float lv[SPD];
  __shared__ float cs_c[12], cs_s[12];
  __shared__ int   cs_p[12], cs_q[12];
  __shared__ float trace;

  const int b = blockIdx.x;
  const int tid = threadIdx.x;
  const float* base = acc + (size_t)b * NPART * ACC_STRIDE;
  const float n = (float)TDIM;

  if (tid < SPD) {
    float s = 0.f;
    for (int p = 0; p < NPART; ++p) s += base[p * ACC_STRIDE + OUT_TRI + tid];
    sv[tid] = s;
  }
  if (tid < SPD * SPD) {
    int i = tid / SPD, j = tid % SPD;
    V[i][j] = (i == j) ? 1.f : 0.f;
  }
  if (tid + 288 < SPD * SPD) {
    int i = (tid + 288) / SPD, j = (tid + 288) % SPD;
    V[i][j] = (i == j) ? 1.f : 0.f;
  }
  __syncthreads();
  for (int e = tid; e < OUT_TRI; e += 288) {
    int i, j;
    triu_ij(e, i, j);
    float g = 0.f;
    for (int p = 0; p < NPART; ++p) g += base[p * ACC_STRIDE + e];
    float v = (g - sv[i] * sv[j] / n) / (n - 1.f + 1e-6f);
    A[i][j] = v;
    A[j][i] = v;   // exactly symmetric -> reference's symmetrize is identity
  }
  __syncthreads();
  if (tid == 0) {
    float t = 0.f;
    for (int i = 0; i < SPD; ++i) t += A[i][i];
    trace = t;
  }
  __syncthreads();
  const float inv = 1.f / (trace + 1e-6f);
  for (int e = tid; e < SPD * SPD; e += 288) {
    int i = e / SPD, j = e % SPD;
    A[i][j] = A[i][j] * inv + ((i == j) ? 1e-4f : 0.f);
  }
  __syncthreads();

  const int pi = tid / SPD;   // pair index 0..11
  const int pj = tid % SPD;   // row/col index 0..23
  for (int sweep = 0; sweep < 12; ++sweep) {
    for (int r = 0; r < SPD - 1; ++r) {
      if (tid < 12) {
        int p, q;
        if (tid == 0) { p = SPD - 1; q = r; }
        else { p = (r + tid) % (SPD - 1); q = (r - tid + (SPD - 1)) % (SPD - 1); }
        if (p > q) { int t = p; p = q; q = t; }
        float apq = A[p][q];
        float c = 1.f, s = 0.f;
        if (fabsf(apq) > 1e-20f) {
          float tau = (A[q][q] - A[p][p]) / (2.f * apq);
          float tt = (tau >= 0.f) ? 1.f / (tau + sqrtf(1.f + tau * tau))
                                  : 1.f / (tau - sqrtf(1.f + tau * tau));
          c = rsqrtf(1.f + tt * tt);
          s = tt * c;
        }
        cs_p[tid] = p; cs_q[tid] = q; cs_c[tid] = c; cs_s[tid] = s;
      }
      __syncthreads();
      {   // left rotations: rows p,q (disjoint across the 12 pairs)
        int p = cs_p[pi], q = cs_q[pi];
        float c = cs_c[pi], s = cs_s[pi];
        float apj = A[p][pj], aqj = A[q][pj];
        A[p][pj] = c * apj - s * aqj;
        A[q][pj] = s * apj + c * aqj;
      }
      __syncthreads();
      {   // right rotations: cols p,q + eigenvector accumulation
        int p = cs_p[pi], q = cs_q[pi];
        float c = cs_c[pi], s = cs_s[pi];
        float ajp = A[pj][p], ajq = A[pj][q];
        A[pj][p] = c * ajp - s * ajq;
        A[pj][q] = s * ajp + c * ajq;
        float vjp = V[pj][p], vjq = V[pj][q];
        V[pj][p] = c * vjp - s * vjq;
        V[pj][q] = s * vjp + c * vjq;
      }
      __syncthreads();
    }
  }

  if (tid < SPD) lv[tid] = logf(fmaxf(A[tid][tid], 1e-6f));
  __syncthreads();
  for (int t = tid; t < OUT_TRI; t += 288) {
    int i, j;
    triu_ij(t, i, j);
    float sum = 0.f;
    for (int k = 0; k < SPD; ++k) sum += V[i][k] * lv[k] * V[j][k];
    out[(size_t)b * OUT_TRI + t] = sum;
  }
}

extern "C" void kernel_launch(void* const* d_in, const int* in_sizes, int n_in,
                              void* d_out, int out_size, void* d_ws, size_t ws_size,
                              hipStream_t stream) {
  (void)in_sizes; (void)n_in; (void)out_size; (void)ws_size;
  const float* x = (const float*)d_in[0];
  const float* W = (const float*)d_in[1];
  float* out = (float*)d_out;

  const size_t ACC_BYTES = (size_t)BATCH * NPART * ACC_STRIDE * sizeof(float);
  float* acc = (float*)d_ws;
  unsigned* Whi = (unsigned*)((char*)d_ws + ACC_BYTES);
  unsigned* Wlo = Whi + WPAD * (DDIM / 2);

  soc_prep_w<<<(WPAD * (DDIM / 2) + 255) / 256, 256, 0, stream>>>(W, Whi, Wlo);
  soc_proj_cov<<<dim3(NPART, BATCH), 256, 0, stream>>>(x, Whi, Wlo, acc);
  soc_finalize<<<BATCH, 288, 0, stream>>>(acc, out);
}
